// GATModel_36825049596097
// MI455X (gfx1250) — matrix operand order repaired and checked by reading, not compiled, not run on hardware
//
#include <hip/hip_runtime.h>
#include <hip/hip_bf16.h>
#include <math.h>

typedef __attribute__((ext_vector_type(16))) _Float16 v16h;
typedef __attribute__((ext_vector_type(4)))  _Float16 h4;
typedef __attribute__((ext_vector_type(8)))  float    v8f;
typedef __attribute__((ext_vector_type(2)))  float    v2f;

static constexpr int cBS = 8, cNV = 8, cMF = 32, cMT = 32, cNP = 4, cKk = 3, cNR = 12;
static constexpr int cHD = 64, cNOUT = 128, cNFEAT = 5000;
static constexpr int BVF = cBS * cNV * cMF;   // 2048
static constexpr int TP  = cMT * cNP;         // 128
static constexpr int KSPLIT = 8;              // K-partitions for GEMM1
static constexpr int KCHUNK = 640;            // 7 x 640 + 520 = 5000
static constexpr int FESZ = BVF * cHD;        // 131072

__device__ __forceinline__ float sigf(float x) { return 1.0f / (1.0f + __expf(-x)); }

__device__ __forceinline__ v8f v8f_zero() {
  v8f z;
#pragma unroll
  for (int i = 0; i < 8; ++i) z[i] = 0.0f;
  return z;
}

// ---------------------------------------------------------------------------
// Prep: transposed diag of relation matrices, h_time projections hs/hc
// ---------------------------------------------------------------------------
__global__ __launch_bounds__(256) void k_prep(const float* __restrict__ params,
                                              const float* __restrict__ ht,
                                              const float* __restrict__ Wp,
                                              const float* __restrict__ cw,
                                              float* __restrict__ diagWT,
                                              float* __restrict__ hs,
                                              float* __restrict__ hc) {
  const int tid = threadIdx.x;
  for (int e = tid; e < cNR * cHD; e += 256) {
    int r = e / cHD, d = e % cHD;
    diagWT[d * cNR + r] = params[((size_t)r * cHD + d) * cHD + d];
  }
  if (tid < cBS * cNV) {
    float s1 = 0.0f, s2 = 0.0f;
    for (int o = 0; o < cNOUT; ++o) {
      float h = ht[(size_t)tid * cNOUT + o];
      s1 += h * Wp[o];
      s2 += h * cw[o];
    }
    hs[tid] = s1;
    hc[tid] = s2;
  }
}

// ---------------------------------------------------------------------------
// GEMM1: feature_embed[2048,64] = feat[2048,5000] @ embeds[5000,64]
// f16 WMMA 16x16x32, f32 accumulate. Block = 4 waves, 64x64 C tile,
// K split 8 ways across blockIdx.y (256 blocks total) into fe_part.
// ---------------------------------------------------------------------------
__global__ __launch_bounds__(128) void k_feat_gemm(const float* __restrict__ feat,
                                                   const float* __restrict__ emb,
                                                   float* __restrict__ fe_part) {
  __shared__ _Float16 Ah[64 * 32];   // [row][k]
  __shared__ _Float16 Bh[64 * 32];   // transposed: [n][k]
  const int tid  = threadIdx.x;
  const int wv   = tid >> 5;
  const int lane = tid & 31;
  const int hf   = (lane >> 4) & 1;
  const int lm   = lane & 15;
  const int row0 = blockIdx.x * 64;
  const int ksp  = blockIdx.y;
  const int kbeg = ksp * KCHUNK;
  const int kend = (kbeg + KCHUNK < cNFEAT) ? kbeg + KCHUNK : cNFEAT;

  v8f acc[4];
#pragma unroll
  for (int nt = 0; nt < 4; ++nt) acc[nt] = v8f_zero();

  for (int kk = kbeg; kk < kend; kk += 32) {
    if (kk + 32 < kend)
      __builtin_prefetch(&feat[(size_t)(row0 + (tid >> 1)) * cNFEAT + kk + 32], 0, 1);

    if (kk + 32 <= kend) {
      // full slab: unconditional vector loads
#pragma unroll
      for (int i = 0; i < 4; ++i) {
        int idx = i * 128 + tid;            // 0..511
        int r = idx >> 3, q = idx & 7;      // A: 64 rows x 8 quads
        float4 va = *(const float4*)(feat + (size_t)(row0 + r) * cNFEAT + kk + q * 4);
        h4 ha = {(_Float16)va.x, (_Float16)va.y, (_Float16)va.z, (_Float16)va.w};
        *(h4*)(&Ah[r * 32 + q * 4]) = ha;
        int kb = idx >> 4, qb = idx & 15;   // B: 32 k x 16 quads of n
        float4 vb = *(const float4*)(emb + (size_t)(kk + kb) * cHD + qb * 4);
        Bh[(qb * 4 + 0) * 32 + kb] = (_Float16)vb.x;
        Bh[(qb * 4 + 1) * 32 + kb] = (_Float16)vb.y;
        Bh[(qb * 4 + 2) * 32 + kb] = (_Float16)vb.z;
        Bh[(qb * 4 + 3) * 32 + kb] = (_Float16)vb.w;
      }
    } else {
      // tail slab (only last split: kk = 4992, 8 valid k)
#pragma unroll
      for (int i = 0; i < 4; ++i) {
        int idx = i * 128 + tid;
        int r = idx >> 3, q = idx & 7;
        float4 va = {0.0f, 0.0f, 0.0f, 0.0f};
        if (q < 2) va = *(const float4*)(feat + (size_t)(row0 + r) * cNFEAT + kk + q * 4);
        h4 ha = {(_Float16)va.x, (_Float16)va.y, (_Float16)va.z, (_Float16)va.w};
        *(h4*)(&Ah[r * 32 + q * 4]) = ha;
        int kb = idx >> 4, qb = idx & 15;
        float4 vb = {0.0f, 0.0f, 0.0f, 0.0f};
        if (kb < 8) vb = *(const float4*)(emb + (size_t)(kk + kb) * cHD + qb * 4);
        Bh[(qb * 4 + 0) * 32 + kb] = (_Float16)vb.x;
        Bh[(qb * 4 + 1) * 32 + kb] = (_Float16)vb.y;
        Bh[(qb * 4 + 2) * 32 + kb] = (_Float16)vb.z;
        Bh[(qb * 4 + 3) * 32 + kb] = (_Float16)vb.w;
      }
    }
    __syncthreads();

    v16h a;
#pragma unroll
    for (int j = 0; j < 16; ++j) {
      int k = (j < 8 ? j : j + 8) + hf * 8;   // 16-bit A layout
      a[j] = Ah[(wv * 16 + lm) * 32 + k];
    }
#pragma unroll
    for (int nt = 0; nt < 4; ++nt) {
      v16h b;
#pragma unroll
      for (int j = 0; j < 16; ++j) b[j] = Bh[(nt * 16 + lm) * 32 + ((j < 8 ? j : j + 8) + hf * 8)];
      acc[nt] = __builtin_amdgcn_wmma_f32_16x16x32_f16(false, a, false, b,
                                                       (short)0, acc[nt], false, false);
    }
    __syncthreads();
  }

  float* dst = fe_part + (size_t)ksp * FESZ;
#pragma unroll
  for (int nt = 0; nt < 4; ++nt)
#pragma unroll
    for (int i = 0; i < 8; ++i) {
      int m = wv * 16 + i + hf * 8;
      int n = nt * 16 + lm;
      dst[(size_t)(row0 + m) * cHD + n] = acc[nt][i];
    }
}

// Deterministic fixed-order reduction of the K-split partials.
__global__ __launch_bounds__(256) void k_reduce(const float* __restrict__ part,
                                                float* __restrict__ fe) {
  int e = blockIdx.x * 256 + threadIdx.x;
  if (e >= FESZ) return;
  float s = 0.0f;
#pragma unroll
  for (int p = 0; p < KSPLIT; ++p) s += part[(size_t)p * FESZ + e];
  fe[e] = s;
}

// ---------------------------------------------------------------------------
// Kernel2: per (b,v,f) block. Async-copy the 18KB rel tile into LDS
// (ASYNCcnt path), WMMA f32 16x16x4 for [128x12]@[12x64] per k-slab with the
// 3-way k-product kept in accumulator registers; then msg / argmax /
// path_attn / (t,p)-contraction.
// ---------------------------------------------------------------------------
__global__ __launch_bounds__(256) void k_message(const float* __restrict__ rel,
                                                 const float* __restrict__ fe,
                                                 const float* __restrict__ diagWT,
                                                 const float* __restrict__ hs,
                                                 const float* __restrict__ Wp,
                                                 const float* __restrict__ path_w,
                                                 const float* __restrict__ path_b,
                                                 const float* __restrict__ gn,
                                                 float* __restrict__ out_pa,
                                                 float* __restrict__ Mj2) {
  __shared__ float RelS[TP * cKk * cNR];   // 4608 floats = 18 KB
  __shared__ float P[TP * cHD];            // Mj tile [128][64] = 32 KB
  __shared__ float lg[TP];
  __shared__ float pa[TP];
  __shared__ int   phat[cMT];

  const int bvf  = blockIdx.x;
  const int tid  = threadIdx.x;
  const int wv   = tid >> 5;
  const int lane = tid & 31;
  const int hf   = (lane >> 4) & 1;
  const int lm   = lane & 15;
  const float* relb = rel + (size_t)bvf * (TP * cKk * cNR);
  const float* feb  = fe + (size_t)bvf * cHD;

  // Async global->LDS copy of the rel tile: 1152 x b128 transfers.
  {
    const unsigned relLds = (unsigned)(size_t)&RelS[0];  // low 32 bits = LDS offset
#pragma unroll
    for (int i = 0; i < 5; ++i) {
      int e = i * 256 + tid;
      if (e < (TP * cKk * cNR) / 4) {
        asm volatile("global_load_async_to_lds_b128 %0, %1, off"
                     :: "v"(relLds + (unsigned)e * 16u), "v"(relb + (size_t)e * 4)
                     : "memory");
      }
    }
    asm volatile("s_wait_asynccnt 0x0" ::: "memory");
  }
  __syncthreads();

  // Preload all A fragments (9 float2 per lane) and run the WMMA chain.
  v2f aF[cKk][3];
#pragma unroll
  for (int ks = 0; ks < cKk; ++ks)
#pragma unroll
    for (int kq = 0; kq < 3; ++kq) {
      const int r0 = kq * 4 + hf * 2;
      const float2 t2 = *(const float2*)&RelS[(((wv * 16 + lm) * cKk) + ks) * cNR + r0];
      aF[ks][kq][0] = t2.x;
      aF[ks][kq][1] = t2.y;
    }

#pragma unroll
  for (int nt = 0; nt < 4; ++nt) {
    v2f bF[3];
#pragma unroll
    for (int kq = 0; kq < 3; ++kq) {
      const int r0 = kq * 4 + hf * 2;
      const float2 t2 = *(const float2*)&diagWT[(nt * 16 + lm) * cNR + r0];
      bF[kq][0] = t2.x;
      bF[kq][1] = t2.y;
    }
    v8f prod = v8f_zero();
#pragma unroll
    for (int ks = 0; ks < cKk; ++ks) {
      v8f accv = v8f_zero();
#pragma unroll
      for (int kq = 0; kq < 3; ++kq)
        accv = __builtin_amdgcn_wmma_f32_16x16x4_f32(false, aF[ks][kq], false, bF[kq],
                                                     (short)0, accv, false, false);
      prod = (ks == 0) ? accv : prod * accv;
    }
#pragma unroll
    for (int i = 0; i < 8; ++i) {
      int tp = wv * 16 + i + hf * 8;
      int d  = nt * 16 + lm;
      P[tp * cHD + d] = feb[d] * prod[i];
    }
  }
  __syncthreads();

  // logits = hs + Mj . Wp[128:] + gumbel
  const float hsv = hs[bvf >> 5];
  if (tid < TP) {
    const float* Wp2 = Wp + cNOUT;
    float s = 0.0f;
    for (int d = 0; d < cHD; ++d) s += P[tid * cHD + d] * Wp2[d];
    lg[tid] = hsv + s + gn[(size_t)bvf * TP + tid];
  }
  __syncthreads();
  if (tid < cMT) {
    int base = tid * cNP;
    int best = 0;
    float bm = lg[base];
    for (int p = 1; p < cNP; ++p) {
      float v = lg[base + p];
      if (v > bm) { bm = v; best = p; }
    }
    phat[tid] = best;
  }
  __syncthreads();
  if (tid < TP) {
    int t = tid >> 2, p = tid & 3;
    float dv = 0.0f;
    if (p == phat[t]) {
      for (int d = 0; d < cHD; ++d) dv += P[tid * cHD + d] * path_w[d];
    }
    float v = sigf(dv + path_b[0]);
    pa[tid] = v;
    out_pa[(size_t)bvf * TP + tid] = v;
  }
  __syncthreads();
  if (tid < cHD) {
    float s = 0.0f;
    for (int tp = 0; tp < TP; ++tp) s += pa[tp] * P[tp * cHD + tid];
    Mj2[(size_t)bvf * cHD + tid] = s;
  }
}

// ---------------------------------------------------------------------------
// Kernel3: per (b,v) normalization / diag trick / causal attention / g_c, g_t
// ---------------------------------------------------------------------------
__global__ __launch_bounds__(64) void k_causal(const float* __restrict__ Mj2,
                                               const float* __restrict__ hc,
                                               const float* __restrict__ cw,
                                               const float* __restrict__ cb,
                                               const float* __restrict__ cthr,
                                               float* __restrict__ out_attn,
                                               float* __restrict__ gC,
                                               float* __restrict__ gT) {
  __shared__ float M[cMF * cHD];
  __shared__ float nrm[cMF], dg[cMF], at[cMF];
  const int bv = blockIdx.x, tid = threadIdx.x;
  for (int e = tid; e < cMF * cHD; e += 64) M[e] = Mj2[(size_t)bv * cMF * cHD + e];
  __syncthreads();
  if (tid < cMF) {
    float s = 0.0f;
    for (int d = 0; d < cHD; ++d) { float x = M[tid * cHD + d]; s += x * x; }
    nrm[tid] = fmaxf(sqrtf(s), 1e-12f);
  }
  __syncthreads();
  if (tid < cMF) {
    float s = 0.0f;
    for (int d = 0; d < cHD; ++d) {
      int x2 = d * cMF + tid;                 // flat-view reshape index
      s += (M[tid * cHD + d] / nrm[tid]) * (M[x2] / nrm[x2 >> 6]);
    }
    dg[tid] = s;
  }
  __syncthreads();
  if (tid < cMF) {
    const float* cw2 = cw + cNOUT;
    float s = 0.0f;
    for (int d = 0; d < cHD; ++d) s += dg[tid] * M[tid * cHD + d] * cw2[d];
    float a = sigf(hc[bv] + s + cb[0] - cthr[0]);
    at[tid] = a;
    out_attn[(size_t)bv * cMF + tid] = a;
  }
  __syncthreads();
  {
    int d = tid;
    float sc = 0.0f, st = 0.0f;
    for (int f = 0; f < cMF; ++f) {
      float m3 = dg[f] * M[f * cHD + d];
      sc += at[f] * m3;
      st += (1.0f - at[f]) * m3;
    }
    gC[(size_t)bv * cHD + d] = sc;
    gT[(size_t)bv * cHD + d] = st;
  }
}

// ---------------------------------------------------------------------------
// Kernel4: three output heads, mean over visits
// ---------------------------------------------------------------------------
__global__ __launch_bounds__(256) void k_out(const float* __restrict__ gC,
                                             const float* __restrict__ gT,
                                             const int* __restrict__ II,
                                             const float* __restrict__ outw,
                                             const float* __restrict__ outb,
                                             float* __restrict__ out) {
  int gid = blockIdx.x * 256 + threadIdx.x;
  if (gid >= cBS * cNOUT) return;
  int b = gid / cNOUT, o = gid % cNOUT;
  int ib = II[b];
  float ai = 0.0f, ac = 0.0f, atv = 0.0f;
  float bo = outb[o];
  for (int v = 0; v < cNV; ++v) {
    const float* gc  = gC + (size_t)(b * cNV + v) * cHD;
    const float* gt  = gT + (size_t)(b * cNV + v) * cHD;
    const float* gti = gT + (size_t)(ib * cNV + v) * cHD;
    float di = 0.0f, dc = 0.0f, dt = 0.0f;
    for (int d = 0; d < cHD; ++d) {
      float w = outw[o * cHD + d];
      dc += gc[d] * w;
      dt += gt[d] * w;
      di += (gc[d] + gti[d]) * w;
    }
    ai += sigf(di + bo);
    ac += sigf(dc + bo);
    atv += sigf(dt + bo);
  }
  out[gid]                   = ai * (1.0f / cNV);
  out[cBS * cNOUT + gid]     = ac * (1.0f / cNV);
  out[2 * cBS * cNOUT + gid] = atv * (1.0f / cNV);
}

// ---------------------------------------------------------------------------
extern "C" void kernel_launch(void* const* d_in, const int* in_sizes, int n_in,
                              void* d_out, int out_size, void* d_ws, size_t ws_size,
                              hipStream_t stream) {
  (void)in_sizes; (void)n_in; (void)out_size; (void)ws_size;
  const float* rel    = (const float*)d_in[0];
  const float* feat   = (const float*)d_in[1];
  const float* h_time = (const float*)d_in[2];
  const float* emb    = (const float*)d_in[3];
  const float* params = (const float*)d_in[4];
  const float* Wp     = (const float*)d_in[5];
  const float* path_w = (const float*)d_in[6];
  const float* path_b = (const float*)d_in[7];
  const float* cw     = (const float*)d_in[8];
  const float* cb     = (const float*)d_in[9];
  const float* cthr   = (const float*)d_in[10];
  const float* outw   = (const float*)d_in[11];
  const float* outb   = (const float*)d_in[12];
  const float* gn     = (const float*)d_in[13];
  const int*   II     = (const int*)d_in[14];

  float* ws      = (float*)d_ws;
  float* fe_part = ws;                               // 8 * 131072
  float* fe      = ws + (size_t)KSPLIT * FESZ;       // 131072
  float* Mj2     = fe + FESZ;                        // 131072
  float* diagWT  = Mj2 + FESZ;                       // 64*12
  float* hs      = diagWT + cNR * cHD;               // 64
  float* hc      = hs + cBS * cNV;                   // 64
  float* gC      = hc + cBS * cNV;                   // 64*64
  float* gT      = gC + cBS * cNV * cHD;             // 64*64

  float* out    = (float*)d_out;
  float* out_pa = out + 3 * cBS * cNOUT;       // path_attn  [2048*128]
  float* out_ac = out_pa + (size_t)BVF * TP;   // attn_causal [64*32]

  k_prep<<<1, 256, 0, stream>>>(params, h_time, Wp, cw, diagWT, hs, hc);
  k_feat_gemm<<<dim3(BVF / 64, KSPLIT), 128, 0, stream>>>(feat, emb, fe_part);
  k_reduce<<<(FESZ + 255) / 256, 256, 0, stream>>>(fe_part, fe);
  k_message<<<BVF, 256, 0, stream>>>(rel, fe, diagWT, hs, Wp, path_w, path_b, gn,
                                     out_pa, Mj2);
  k_causal<<<cBS * cNV, 64, 0, stream>>>(Mj2, hc, cw, cb, cthr, out_ac, gC, gT);
  k_out<<<(cBS * cNOUT + 255) / 256, 256, 0, stream>>>(gC, gT, II, outw, outb, out);
}
